// TransformerEncoder_45157286150152
// MI455X (gfx1250) — matrix operand order repaired and checked by reading, not compiled, not run on hardware
//
#include <hip/hip_runtime.h>
#include <hip/hip_bf16.h>
#include <math.h>

// Problem constants (match reference)
#define BB   64
#define LL   128
#define DD   512
#define HH   8
#define DHH  64
#define FFD  1024
#define NTT  10000
#define TT   65536
#define NL   2
#define ROWS (BB * LL)  // 8192

typedef __attribute__((ext_vector_type(16))) _Float16 v16h;
typedef __attribute__((ext_vector_type(8)))  _Float16 v8h;
typedef __attribute__((ext_vector_type(8)))  float    v8f;

static __device__ __forceinline__ v16h cat16(v8h a, v8h b) {
  v16h r;
#pragma unroll
  for (int i = 0; i < 8; ++i) { r[i] = a[i]; r[i + 8] = b[i]; }
  return r;
}
static __device__ __forceinline__ v8h ld8(const _Float16* p) {
  return *(const v8h*)p;  // 16B aligned by construction everywhere below
}

// ---------------------------------------------------------------------------
// f32 -> f16 weight conversion + zero fill
// ---------------------------------------------------------------------------
__global__ void f32_to_f16(const float* __restrict__ src, _Float16* __restrict__ dst, int n) {
  int i = blockIdx.x * blockDim.x + threadIdx.x;
  int stride = gridDim.x * blockDim.x;
  for (; i < n; i += stride) dst[i] = (_Float16)src[i];
}
__global__ void zero_f16(_Float16* __restrict__ dst, int n) {
  int i = blockIdx.x * blockDim.x + threadIdx.x;
  if (i < n) dst[i] = (_Float16)0.0f;
}

// ---------------------------------------------------------------------------
// Embedding gather + segment mean + sinusoidal positional encoding
// One block per output row (B*L rows). seg_ids is sorted -> binary search.
// ---------------------------------------------------------------------------
__global__ __launch_bounds__(256)
void embed_kernel(const int* __restrict__ item_ids, const int* __restrict__ seg_ids,
                  const float* __restrict__ emb, float* __restrict__ x,
                  _Float16* __restrict__ xh) {
  const int row = blockIdx.x;
  int lo = 0, hi = TT;
  while (lo < hi) { int mid = (lo + hi) >> 1; if (seg_ids[mid] < row) lo = mid + 1; else hi = mid; }
  const int start = lo;
  hi = TT;
  while (lo < hi) { int mid = (lo + hi) >> 1; if (seg_ids[mid] <= row) lo = mid + 1; else hi = mid; }
  const int end = lo;
  const float cnt = (float)(end - start);
  const int l = row & (LL - 1);
  const float kf = logf(10000.0f) / (float)DD;
  for (int d = threadIdx.x; d < DD; d += 256) {
    float sum = 0.0f;
    for (int j = start; j < end; ++j) sum += emb[(size_t)item_ids[j] * DD + d];
    float mean = (cnt > 0.0f) ? sum / cnt : 0.0f;
    float den = __expf(-(float)(d & ~1) * kf);
    float ang = (float)l * den;
    float pe = (d & 1) ? cosf(ang) : sinf(ang);
    float val = mean + pe;
    x[(size_t)row * DD + d] = val;
    xh[(size_t)row * DD + d] = (_Float16)val;
  }
}

// ---------------------------------------------------------------------------
// WMMA GEMM: C[M,N] = A[M,K] @ W[N,K]^T + bias[N]   (f16 in, f32 acc)
// Block = 256 threads = 8 waves (4 m-rows x 2 n-cols); block tile 128x128;
// wave tile 32x64 (2x4 WMMA tiles). B tile is staged into LDS with
// double-buffered global_load_async_to_lds_b128 (ASYNCcnt pipeline).
// Out-of-range N rows are staged from a zero-filled scratch (zpad), so the
// inner loop needs no masking; bias-init and stores keep their n<N guards.
// M must be a multiple of 128, K a multiple of 32; N is bounds-checked.
// ---------------------------------------------------------------------------
#define BROW_P 40  // padded LDS row stride in halves (80B = 20 dwords; 16 conflict-free banks)

__global__ __launch_bounds__(256)
void gemm_f16(const _Float16* __restrict__ A, const _Float16* __restrict__ W,
              const float* __restrict__ bias, const _Float16* __restrict__ zpad,
              int M, int N, int K,
              float* __restrict__ outF, _Float16* __restrict__ outH, int relu) {
  (void)M;
  __shared__ _Float16 bufB[2][128 * BROW_P];

  const int tid  = threadIdx.x;
  const int lane = tid & 31;
  const int w    = tid >> 5;
  const int lo16 = lane & 15;
  const int hi   = lane >> 4;
  const int wr = w >> 1, wc = w & 1;
  const int m0    = blockIdx.y * 128 + wr * 32;
  const int nBase = blockIdx.x * 128;
  const int n0    = nBase + wc * 64;

  // ---- async staging setup: each thread copies 2 x 16B chunks per K-step.
  // OOB rows read from the zero scratch -> staged tile is already masked.
  const int rrow = tid >> 2;  // 0..63
  const int ch   = tid & 3;   // 16B chunk within the 64B row slice
  const _Float16* srcRow[2];
  uint32_t dstOff[2];
  {
    int gn0 = nBase + rrow, gn1 = nBase + rrow + 64;
    srcRow[0] = (gn0 < N) ? (W + (size_t)gn0 * K + ch * 8) : (zpad + ch * 8);
    srcRow[1] = (gn1 < N) ? (W + (size_t)gn1 * K + ch * 8) : (zpad + ch * 8);
    dstOff[0] = (uint32_t)(uintptr_t)&bufB[0][rrow * BROW_P + ch * 8];
    dstOff[1] = (uint32_t)(uintptr_t)&bufB[0][(rrow + 64) * BROW_P + ch * 8];
  }
  const uint32_t bufStride =
      (uint32_t)((uintptr_t)&bufB[1][0] - (uintptr_t)&bufB[0][0]);

  auto stage = [&](int k0, int buf) {
    uint64_t g0 = (uint64_t)(uintptr_t)(srcRow[0] + k0);
    uint64_t g1 = (uint64_t)(uintptr_t)(srcRow[1] + k0);
    uint32_t d0 = dstOff[0] + (uint32_t)buf * bufStride;
    uint32_t d1 = dstOff[1] + (uint32_t)buf * bufStride;
    asm volatile("global_load_async_to_lds_b128 %0, %1, off"
                 :: "v"(d0), "v"(g0) : "memory");
    asm volatile("global_load_async_to_lds_b128 %0, %1, off"
                 :: "v"(d1), "v"(g1) : "memory");
  };

  // ---- accumulators initialized with bias (C/D layout: lane lo16 = column)
  v8f acc[2][4];
#pragma unroll
  for (int nt = 0; nt < 4; ++nt) {
    int n = n0 + nt * 16 + lo16;
    float bv = (n < N) ? bias[n] : 0.0f;
#pragma unroll
    for (int mt = 0; mt < 2; ++mt)
#pragma unroll
      for (int r = 0; r < 8; ++r) acc[mt][nt][r] = bv;
  }

  const _Float16* arow[2] = { A + (size_t)(m0 + lo16) * K,
                              A + (size_t)(m0 + 16 + lo16) * K };

  stage(0, 0);
  int cur = 0;
  for (int k0 = 0; k0 < K; k0 += 32) {
    const bool more = (k0 + 32) < K;
    if (more) {
      stage(k0 + 32, cur ^ 1);
      __builtin_prefetch(arow[0] + k0 + 64, 0, 3);
      asm volatile("s_wait_asynccnt 0x2" ::: "memory");
    } else {
      asm volatile("s_wait_asynccnt 0x0" ::: "memory");
    }
    __syncthreads();  // B tile for this K-step visible to all waves

    v16h af[2];  // A layout: halves {k0+8hi..}, {k0+16+8hi..}
    af[0] = cat16(ld8(arow[0] + k0 + 8 * hi), ld8(arow[0] + k0 + 16 + 8 * hi));
    af[1] = cat16(ld8(arow[1] + k0 + 8 * hi), ld8(arow[1] + k0 + 16 + 8 * hi));

    const _Float16* bb = &bufB[cur][0];
#pragma unroll
    for (int nt = 0; nt < 4; ++nt) {  // B layout: halves {8hi..}, {16+8hi..} (matches A)
      const _Float16* brow = bb + (wc * 64 + nt * 16 + lo16) * BROW_P + 8 * hi;
      v16h bf = cat16(ld8(brow), ld8(brow + 16));
      acc[0][nt] = __builtin_amdgcn_wmma_f32_16x16x32_f16(
          false, af[0], false, bf, (short)0, acc[0][nt], false, false);
      acc[1][nt] = __builtin_amdgcn_wmma_f32_16x16x32_f16(
          false, af[1], false, bf, (short)0, acc[1][nt], false, false);
    }
    __syncthreads();  // all waves done with bufB[cur] before it is re-staged
    cur ^= 1;
  }

#pragma unroll
  for (int mt = 0; mt < 2; ++mt)
#pragma unroll
    for (int nt = 0; nt < 4; ++nt) {
      int n = n0 + nt * 16 + lo16;
      if (n >= N) continue;
#pragma unroll
      for (int r = 0; r < 8; ++r) {
        int m = m0 + mt * 16 + r + 8 * hi;
        float val = acc[mt][nt][r];
        if (relu) val = fmaxf(val, 0.0f);
        if (outH) outH[(size_t)m * N + n] = (_Float16)val;
        else      outF[(size_t)m * N + n] = val;
      }
    }
}

// ---------------------------------------------------------------------------
// Attention: one workgroup per (b,h). scores=q@k^T (WMMA), softmax, o=attn@v
// (WMMA, with v staged transposed in LDS). Writes o with heads concatenated.
// ---------------------------------------------------------------------------
__global__ __launch_bounds__(256)
void attn_kernel(const _Float16* __restrict__ qkvh, const float* __restrict__ fmask,
                 const unsigned char* __restrict__ pmask, _Float16* __restrict__ oh) {
  __shared__ _Float16 vT[DHH][LL + 8];      // v transposed: [dh][l]
  __shared__ _Float16 attnS[LL][LL + 8];    // softmaxed attention, f16
  const int h = blockIdx.x;
  const int b = blockIdx.y;
  const int tid = threadIdx.x;

  // stage v -> vT (ds writes)
  for (int idx = tid; idx < LL * DHH; idx += 256) {
    int dh = idx & (DHH - 1);
    int l  = idx >> 6;
    vT[dh][l] = qkvh[(size_t)(b * LL + l) * (3 * DD) + 2 * DD + h * DHH + dh];
  }
  __syncthreads();

  const int lane = tid & 31, w = tid >> 5;
  const int lo16 = lane & 15, hi = lane >> 4;
  const int m0 = w * 16;  // each wave owns a 16-row strip of the 128x128 score tile

  v8f accS[8] = {};
  const _Float16* qrow = qkvh + (size_t)(b * LL + m0 + lo16) * (3 * DD) + h * DHH;
#pragma unroll
  for (int k0 = 0; k0 < DHH; k0 += 32) {
    v16h a = cat16(ld8(qrow + k0 + 8 * hi), ld8(qrow + k0 + 16 + 8 * hi));
#pragma unroll
    for (int nt = 0; nt < 8; ++nt) {
      const _Float16* krow = qkvh + (size_t)(b * LL + nt * 16 + lo16) * (3 * DD)
                           + DD + h * DHH + k0 + 8 * hi;
      v16h bf = cat16(ld8(krow), ld8(krow + 16));
      accS[nt] = __builtin_amdgcn_wmma_f32_16x16x32_f16(
          false, a, false, bf, (short)0, accS[nt], false, false);
    }
  }

  // scale + masks + row softmax (rows live in 16-lane groups of the C layout)
  const float scale = 0.125f;  // 1/sqrt(64)
#pragma unroll
  for (int r = 0; r < 8; ++r) {
    const int row = m0 + r + 8 * hi;
    float mx = -3.4e38f;
#pragma unroll
    for (int nt = 0; nt < 8; ++nt) {
      int col = nt * 16 + lo16;
      float sc = accS[nt][r] * scale + fmask[row * LL + col]
               + (pmask[b * LL + col] ? -1e9f : 0.0f);
      accS[nt][r] = sc;
      mx = fmaxf(mx, sc);
    }
#pragma unroll
    for (int off = 1; off < 16; off <<= 1) mx = fmaxf(mx, __shfl_xor(mx, off, 32));
    float sum = 0.0f;
#pragma unroll
    for (int nt = 0; nt < 8; ++nt) {
      float e = __expf(accS[nt][r] - mx);
      accS[nt][r] = e;
      sum += e;
    }
#pragma unroll
    for (int off = 1; off < 16; off <<= 1) sum += __shfl_xor(sum, off, 32);
    float inv = 1.0f / sum;
#pragma unroll
    for (int nt = 0; nt < 8; ++nt)
      attnS[row][nt * 16 + lo16] = (_Float16)(accS[nt][r] * inv);
  }
  __syncthreads();

  // o = attn @ v   (A from LDS attnS, B from LDS vT)
  v8f accO[4] = {};
  const _Float16* arow2 = &attnS[m0 + lo16][0];
#pragma unroll
  for (int k0 = 0; k0 < LL; k0 += 32) {
    v16h a = cat16(ld8(arow2 + k0 + 8 * hi), ld8(arow2 + k0 + 16 + 8 * hi));
#pragma unroll
    for (int nt = 0; nt < 4; ++nt) {
      const _Float16* vrow = &vT[nt * 16 + lo16][k0 + 8 * hi];
      v16h bf = cat16(ld8(vrow), ld8(vrow + 16));
      accO[nt] = __builtin_amdgcn_wmma_f32_16x16x32_f16(
          false, a, false, bf, (short)0, accO[nt], false, false);
    }
  }
#pragma unroll
  for (int nt = 0; nt < 4; ++nt)
#pragma unroll
    for (int r = 0; r < 8; ++r) {
      int row = b * LL + m0 + r + 8 * hi;
      int col = h * DHH + nt * 16 + lo16;
      oh[(size_t)row * DD + col] = (_Float16)accO[nt][r];
    }
}

// ---------------------------------------------------------------------------
// Residual + LayerNorm: x = LN(x + delta); also emits f16 copy.
// One block per row; D=512, 2 elements per thread.
// ---------------------------------------------------------------------------
__global__ __launch_bounds__(256)
void ln_kernel(float* __restrict__ x, const float* __restrict__ delta,
               const float* __restrict__ gam, const float* __restrict__ bet,
               _Float16* __restrict__ xh) {
  __shared__ float red[256];
  const int row = blockIdx.x, tid = threadIdx.x;
  const size_t base = (size_t)row * DD;
  float u0 = x[base + tid]       + delta[base + tid];
  float u1 = x[base + tid + 256] + delta[base + tid + 256];
  red[tid] = u0 + u1;
  __syncthreads();
  for (int off = 128; off > 0; off >>= 1) {
    if (tid < off) red[tid] += red[tid + off];
    __syncthreads();
  }
  float mean = red[0] * (1.0f / DD);
  __syncthreads();
  float d0 = u0 - mean, d1 = u1 - mean;
  red[tid] = d0 * d0 + d1 * d1;
  __syncthreads();
  for (int off = 128; off > 0; off >>= 1) {
    if (tid < off) red[tid] += red[tid + off];
    __syncthreads();
  }
  float rstd = rsqrtf(red[0] * (1.0f / DD) + 1e-5f);
  float y0 = d0 * rstd * gam[tid]       + bet[tid];
  float y1 = d1 * rstd * gam[tid + 256] + bet[tid + 256];
  x[base + tid] = y0;       x[base + tid + 256] = y1;
  xh[base + tid] = (_Float16)y0;  xh[base + tid + 256] = (_Float16)y1;
}

// ---------------------------------------------------------------------------
extern "C" void kernel_launch(void* const* d_in, const int* in_sizes, int n_in,
                              void* d_out, int out_size, void* d_ws, size_t ws_size,
                              hipStream_t stream) {
  (void)in_sizes; (void)n_in; (void)out_size; (void)ws_size;
  const int*   item_ids = (const int*)d_in[0];
  const int*   seg_ids  = (const int*)d_in[1];
  const unsigned char* pad_mask = (const unsigned char*)d_in[2];
  const float* fmask = (const float*)d_in[3];
  // d_in[4] = max_len scalar (L fixed at 128)
  const float* emb  = (const float*)d_in[5];
  const float* Wqkv = (const float*)d_in[6];
  const float* bqkv = (const float*)d_in[7];
  const float* Wo   = (const float*)d_in[8];
  const float* bo   = (const float*)d_in[9];
  const float* W1   = (const float*)d_in[10];
  const float* b1   = (const float*)d_in[11];
  const float* W2   = (const float*)d_in[12];
  const float* b2   = (const float*)d_in[13];
  const float* ln1s = (const float*)d_in[14];
  const float* ln1b = (const float*)d_in[15];
  const float* ln2s = (const float*)d_in[16];
  const float* ln2b = (const float*)d_in[17];
  const float* Wout = (const float*)d_in[18];
  const float* bout = (const float*)d_in[19];

  // workspace carve-up (~106 MB total)
  char* ws = (char*)d_ws;
  size_t off = 0;
  auto carve = [&](size_t bytes) -> char* {
    char* p = ws + off;
    off += (bytes + 255) & ~(size_t)255;
    return p;
  };
  float*    x     = (float*)   carve((size_t)ROWS * DD * 4);
  _Float16* xh    = (_Float16*)carve((size_t)ROWS * DD * 2);
  _Float16* qkvh  = (_Float16*)carve((size_t)ROWS * 3 * DD * 2);
  _Float16* ohb   = (_Float16*)carve((size_t)ROWS * DD * 2);
  float*    delta = (float*)   carve((size_t)ROWS * DD * 4);
  _Float16* h1h   = (_Float16*)carve((size_t)ROWS * FFD * 2);
  _Float16* Wqkvh = (_Float16*)carve((size_t)NL * 3 * DD * DD * 2);
  _Float16* Woh   = (_Float16*)carve((size_t)NL * DD * DD * 2);
  _Float16* W1h   = (_Float16*)carve((size_t)NL * FFD * DD * 2);
  _Float16* W2h   = (_Float16*)carve((size_t)NL * DD * FFD * 2);
  _Float16* Wouth = (_Float16*)carve((size_t)NTT * DD * 2);
  _Float16* zpad  = (_Float16*)carve((size_t)FFD * 2);  // zero scratch, K_max halves

  auto cvt = [&](const float* s, _Float16* d, int n) {
    int blocks = (n + 255) / 256;
    if (blocks > 4096) blocks = 4096;
    f32_to_f16<<<blocks, 256, 0, stream>>>(s, d, n);
  };
  cvt(Wqkv, Wqkvh, NL * 3 * DD * DD);
  cvt(Wo,   Woh,   NL * DD * DD);
  cvt(W1,   W1h,   NL * FFD * DD);
  cvt(W2,   W2h,   NL * DD * FFD);
  cvt(Wout, Wouth, NTT * DD);
  zero_f16<<<(FFD + 255) / 256, 256, 0, stream>>>(zpad, FFD);

  embed_kernel<<<ROWS, 256, 0, stream>>>(item_ids, seg_ids, emb, x, xh);

  for (int i = 0; i < NL; ++i) {
    // qkv = x @ Wqkv^T + bqkv   -> f16 [ROWS, 3D]
    gemm_f16<<<dim3((3 * DD) / 128, ROWS / 128), 256, 0, stream>>>(
        xh, Wqkvh + (size_t)i * 3 * DD * DD, bqkv + (size_t)i * 3 * DD, zpad,
        ROWS, 3 * DD, DD, nullptr, qkvh, 0);
    // attention
    attn_kernel<<<dim3(HH, BB), 256, 0, stream>>>(qkvh, fmask, pad_mask, ohb);
    // o @ Wo^T + bo -> delta (f32)
    gemm_f16<<<dim3(DD / 128, ROWS / 128), 256, 0, stream>>>(
        ohb, Woh + (size_t)i * DD * DD, bo + (size_t)i * DD, zpad,
        ROWS, DD, DD, delta, nullptr, 0);
    // x = LN(x + delta)
    ln_kernel<<<ROWS, 256, 0, stream>>>(x, delta, ln1s + (size_t)i * DD,
                                        ln1b + (size_t)i * DD, xh);
    // h1 = relu(x @ W1^T + b1) -> f16
    gemm_f16<<<dim3(FFD / 128, ROWS / 128), 256, 0, stream>>>(
        xh, W1h + (size_t)i * FFD * DD, b1 + (size_t)i * FFD, zpad,
        ROWS, FFD, DD, nullptr, h1h, 1);
    // ff = h1 @ W2^T + b2 -> delta (f32)
    gemm_f16<<<dim3(DD / 128, ROWS / 128), 256, 0, stream>>>(
        h1h, W2h + (size_t)i * DD * FFD, b2 + (size_t)i * DD, zpad,
        ROWS, DD, FFD, delta, nullptr, 0);
    // x = LN(x + ff)
    ln_kernel<<<ROWS, 256, 0, stream>>>(x, delta, ln2s + (size_t)i * DD,
                                        ln2b + (size_t)i * DD, xh);
  }

  // logits = x @ Wout^T + bout -> d_out (f32 [ROWS, 10000]); N tail handled
  gemm_f16<<<dim3((NTT + 127) / 128, ROWS / 128), 256, 0, stream>>>(
      xh, Wouth, bout, zpad, ROWS, NTT, DD, (float*)d_out, nullptr, 0);
}
